// TabularGNN_73589969649951
// MI455X (gfx1250) — matrix operand order repaired and checked by reading, not compiled, not run on hardware
//
#include <hip/hip_runtime.h>
#include <stdint.h>

// MI455X / gfx1250, wave32. WMMA bf16 16x16x32, f32 accumulate.

typedef __attribute__((ext_vector_type(16))) __bf16 v16bf;
typedef __attribute__((ext_vector_type(8)))  float  v8f;

#define N_PTS 16384
#define DIM   256
#define KNN   10

union V16BF { v16bf v; uint4 q[2]; };

__device__ __forceinline__ unsigned short f2bf(float f) {
  unsigned int u = __float_as_uint(f);
  unsigned int r = u + 0x7FFFu + ((u >> 16) & 1u);   // round-to-nearest-even
  return (unsigned short)(r >> 16);
}

// A-matrix tile (16x32 bf16), ISA layout: lane l<16 -> row (l&15), K = {0..7, 16..23};
// lanes 16..31 -> same rows, K shifted by +8. Two 16B contiguous loads per lane.
__device__ __forceinline__ v16bf load_a_tile(const unsigned short* __restrict__ base,
                                             int row, int kc, int lane) {
  const unsigned short* p = base + row * DIM + kc * 32 + ((lane >> 4) << 3);
  V16BF u;
  u.q[0] = *reinterpret_cast<const uint4*>(p);
  u.q[1] = *reinterpret_cast<const uint4*>(p + 16);
  return u.v;
}

// B-matrix tile (32x16 bf16), lane l -> column (l&15), K = {0..7, 16..23} for low lanes,
// K shifted by +8 for high lanes (matches A). Two 16B contiguous loads per lane.
__device__ __forceinline__ v16bf load_b_tile(const unsigned short* __restrict__ colptr,
                                             int kc, int lane) {
  const unsigned short* p = colptr + kc * 32 + ((lane >> 4) << 3);
  V16BF u;
  u.q[0] = *reinterpret_cast<const uint4*>(p);
  u.q[1] = *reinterpret_cast<const uint4*>(p + 16);
  return u.v;
}

// ---------------- setup kernels ----------------

__global__ void k_convert_sq(const float* __restrict__ x,
                             unsigned short* __restrict__ xb,
                             float* __restrict__ sq) {
  const int row = blockIdx.x, c = threadIdx.x;
  float v = x[row * DIM + c];
  xb[row * DIM + c] = f2bf(v);
  __shared__ float red[256];
  red[c] = v * v;
  __syncthreads();
  for (int s = 128; s > 0; s >>= 1) {
    if (c < s) red[c] += red[c + s];
    __syncthreads();
  }
  if (c == 0) sq[row] = red[0];
}

// W (256x256 f32, k-major) -> bf16 transposed (n-major) so B loads are k-contiguous
__global__ void k_convert_wt(const float* __restrict__ w,
                             unsigned short* __restrict__ wbt) {
  const int idx = blockIdx.x * 256 + threadIdx.x;      // 65536
  const int k = idx >> 8, n = idx & 255;
  wbt[n * 256 + k] = f2bf(w[idx]);
}

// ---------------- fused distance-GEMM + top-K ----------------
// One wave owns 16 rows. Distances computed 16x16 tile at a time via 8 WMMAs
// (K=256). d2 tile staged through per-wave LDS; all 32 lanes do selection:
// lane r keeps top-11 of columns n=0..7, lane r+16 of n=8..15; merged at end.

__global__ void __launch_bounds__(256) k_knn(const unsigned short* __restrict__ xb,
                                             const float* __restrict__ sq,
                                             int* __restrict__ nbr) {
  __shared__ float tile[8][16][17];                 // per-wave 16x16 d2 staging (padded)
  __shared__ float mD[8][16][11];                   // high-half lists for final merge
  __shared__ int   mI[8][16][11];

  const int tid  = threadIdx.x;
  const int wave = tid >> 5;
  const int lane = tid & 31;
  const int half = lane >> 4;                       // 0: lanes 0-15, 1: lanes 16-31
  const int lrow = lane & 15;
  const int rowBase = blockIdx.x * 128 + wave * 16;
  const int myrowA  = rowBase + lrow;
  const int r0 = half << 3;                         // C-layout row offset: 0 or 8
  const int nOff = half << 3;                       // selection column offset: 0 or 8

  // Preload the 16x256 bf16 A strip: 8 k-chunks, register resident.
  v16bf a[8];
#pragma unroll
  for (int kc = 0; kc < 8; ++kc) a[kc] = load_a_tile(xb, myrowA, kc, lane);

  float sqi[8];
#pragma unroll
  for (int r = 0; r < 8; ++r) sqi[r] = sq[rowBase + r0 + r];

  float bestD[11]; int bestI[11];
#pragma unroll
  for (int t = 0; t < 11; ++t) { bestD[t] = 3.0e38f; bestI[t] = 0; }

  for (int ct = 0; ct < N_PTS / 16; ++ct) {
    const int col0  = ct * 16;
    const int mycol = col0 + lrow;
    const unsigned short* bptr = xb + mycol * DIM;
    if (ct + 1 < N_PTS / 16)
      __builtin_prefetch(xb + (mycol + 16) * DIM, 0, 3);   // global_prefetch_b8

    // Preload all 8 B chunks (one clause of b128 loads), then run the WMMA chain
    // so loadcnt drains progressively instead of a full wait per WMMA.
    v16bf b[8];
#pragma unroll
    for (int kc = 0; kc < 8; ++kc) b[kc] = load_b_tile(bptr, kc, lane);

    v8f c = {};
#pragma unroll
    for (int kc = 0; kc < 8; ++kc)
      c = __builtin_amdgcn_wmma_f32_16x16x32_bf16(false, a[kc], false, b[kc],
                                                  (short)0, c, false, false);

    const float sqj = sq[mycol];
#pragma unroll
    for (int r = 0; r < 8; ++r) {
      float d2 = fmaxf(sqi[r] + sqj - 2.0f * c[r], 0.0f);
      tile[wave][r0 + r][lrow] = d2;
    }
    asm volatile("s_wait_dscnt 0" ::: "memory");    // wave-local LDS RAW (DS in-order)

    // All 32 lanes select: 8 candidates each.
#pragma unroll 1
    for (int n = 0; n < 8; ++n) {
      float d = tile[wave][lrow][nOff + n];
      if (d < bestD[10]) {                          // rare once list saturates
        float cd = d; int cj = col0 + nOff + n;
#pragma unroll
        for (int p = 0; p < 11; ++p) {
          if (cd < bestD[p]) {
            float td = bestD[p]; bestD[p] = cd; cd = td;
            int   ti = bestI[p]; bestI[p] = cj; cj = ti;
          }
        }
      }
    }
  }

  // Merge: high lanes publish their lists, low lanes fold them in.
  if (half) {
#pragma unroll
    for (int t = 0; t < 11; ++t) { mD[wave][lrow][t] = bestD[t]; mI[wave][lrow][t] = bestI[t]; }
  }
  asm volatile("s_wait_dscnt 0" ::: "memory");
  if (!half) {
#pragma unroll 1
    for (int t = 0; t < 11; ++t) {
      float d = mD[wave][lrow][t];
      if (d < bestD[10]) {
        float cd = d; int cj = mI[wave][lrow][t];
#pragma unroll
        for (int p = 0; p < 11; ++p) {
          if (cd < bestD[p]) {
            float td = bestD[p]; bestD[p] = cd; cd = td;
            int   ti = bestI[p]; bestI[p] = cj; cj = ti;
          }
        }
      }
    }
    const int row = rowBase + lrow;
#pragma unroll
    for (int t = 0; t < KNN; ++t) nbr[row * KNN + t] = bestI[t + 1];  // drop self
  }
}

// ---------------- degree ----------------

__global__ void k_deg_init(float* __restrict__ deg) {
  deg[blockIdx.x * 256 + threadIdx.x] = 1.0f;
}
__global__ void k_deg_count(const int* __restrict__ nbr, float* __restrict__ deg) {
  const int e = blockIdx.x * 256 + threadIdx.x;
  if (e < N_PTS * KNN) atomicAdd(&deg[nbr[e]], 1.0f);
}
__global__ void k_deg_fin(const float* __restrict__ deg,
                          float* __restrict__ isd, float* __restrict__ invd) {
  const int i = blockIdx.x * 256 + threadIdx.x;
  const float d = deg[i];
  isd[i]  = rsqrtf(d);
  invd[i] = 1.0f / d;
}

// ---------------- layer GEMM: hw = hb(bf16) @ W(bf16,transposed), f32 out ----------------

__global__ void __launch_bounds__(256) k_gemm(const unsigned short* __restrict__ hb,
                                              const unsigned short* __restrict__ wbt,
                                              float* __restrict__ hw) {
  const int tid  = threadIdx.x;
  const int wave = tid >> 5, lane = tid & 31;
  const int rowBase = blockIdx.x * 128 + wave * 16;
  const int myrowA  = rowBase + (lane & 15);
  const int r0 = (lane >> 4) << 3;

  v16bf a[8];
#pragma unroll
  for (int kc = 0; kc < 8; ++kc) a[kc] = load_a_tile(hb, myrowA, kc, lane);

  for (int ct = 0; ct < 16; ++ct) {
    const int n = ct * 16 + (lane & 15);
    const unsigned short* bptr = wbt + n * 256;

    v16bf b[8];
#pragma unroll
    for (int kc = 0; kc < 8; ++kc) b[kc] = load_b_tile(bptr, kc, lane);

    v8f c = {};
#pragma unroll
    for (int kc = 0; kc < 8; ++kc)
      c = __builtin_amdgcn_wmma_f32_16x16x32_bf16(false, a[kc], false, b[kc],
                                                  (short)0, c, false, false);
#pragma unroll
    for (int r = 0; r < 8; ++r) hw[(rowBase + r0 + r) * 256 + n] = c[r];
  }
}

// ---------------- aggregation ----------------

__global__ void k_init_acc(const float* __restrict__ hw, const float* __restrict__ invd,
                           float* __restrict__ acc) {
  const int idx = blockIdx.x * 256 + threadIdx.x;
  acc[idx] = hw[idx] * invd[idx >> 8];                 // self term hw * inv_deg
}

__global__ void k_scatter(const float* __restrict__ hw, const int* __restrict__ nbr,
                          const float* __restrict__ isd, float* __restrict__ acc) {
  const int i = blockIdx.x, c = threadIdx.x;
  const float v  = hw[i * DIM + c];
  const float wi = isd[i];
#pragma unroll 1
  for (int t = 0; t < KNN; ++t) {
    const int j = nbr[i * KNN + t];                    // uniform -> scalar load
    atomicAdd(&acc[j * DIM + c], v * (wi * isd[j]));   // global_atomic_add_f32, L2 resident
  }
}

__global__ void k_finalize_relu(const float* __restrict__ acc, const float* __restrict__ b,
                                unsigned short* __restrict__ hb) {
  const int idx = blockIdx.x * 256 + threadIdx.x;
  float v = fmaxf(acc[idx] + b[idx & 255], 0.0f);
  hb[idx] = f2bf(v);
}
__global__ void k_finalize_out(const float* __restrict__ acc, const float* __restrict__ b,
                               float* __restrict__ out) {
  const int idx = blockIdx.x * 256 + threadIdx.x;
  out[idx] = acc[idx] + b[idx & 255];
}

// ---------------- driver ----------------

extern "C" void kernel_launch(void* const* d_in, const int* in_sizes, int n_in,
                              void* d_out, int out_size, void* d_ws, size_t ws_size,
                              hipStream_t stream) {
  const float* x  = (const float*)d_in[0];
  const float* W1 = (const float*)d_in[1];
  const float* b1 = (const float*)d_in[2];
  const float* W2 = (const float*)d_in[3];
  const float* b2 = (const float*)d_in[4];
  const float* W3 = (const float*)d_in[5];
  const float* b3 = (const float*)d_in[6];
  float* out = (float*)d_out;

  char* ws = (char*)d_ws;
  size_t off = 0;
  auto alloc = [&](size_t bytes) -> void* {
    void* p = ws + off;
    off += (bytes + 255) & ~size_t(255);
    return p;
  };

  unsigned short* xb   = (unsigned short*)alloc((size_t)N_PTS * DIM * 2);
  float*          sq   = (float*)alloc((size_t)N_PTS * 4);
  int*            nbr  = (int*)alloc((size_t)N_PTS * KNN * 4);
  float*          deg  = (float*)alloc((size_t)N_PTS * 4);
  float*          isd  = (float*)alloc((size_t)N_PTS * 4);
  float*          invd = (float*)alloc((size_t)N_PTS * 4);
  unsigned short* wbt1 = (unsigned short*)alloc((size_t)DIM * DIM * 2);
  unsigned short* wbt2 = (unsigned short*)alloc((size_t)DIM * DIM * 2);
  unsigned short* wbt3 = (unsigned short*)alloc((size_t)DIM * DIM * 2);
  float*          hw   = (float*)alloc((size_t)N_PTS * DIM * 4);
  float*          acc  = (float*)alloc((size_t)N_PTS * DIM * 4);
  unsigned short* hb   = (unsigned short*)alloc((size_t)N_PTS * DIM * 2);

  k_convert_sq<<<N_PTS, 256, 0, stream>>>(x, xb, sq);
  k_convert_wt<<<256, 256, 0, stream>>>(W1, wbt1);
  k_convert_wt<<<256, 256, 0, stream>>>(W2, wbt2);
  k_convert_wt<<<256, 256, 0, stream>>>(W3, wbt3);

  k_knn<<<N_PTS / 128, 256, 0, stream>>>(xb, sq, nbr);

  k_deg_init<<<N_PTS / 256, 256, 0, stream>>>(deg);
  k_deg_count<<<(N_PTS * KNN + 255) / 256, 256, 0, stream>>>(nbr, deg);
  k_deg_fin<<<N_PTS / 256, 256, 0, stream>>>(deg, isd, invd);

  // layer 1 (input = bf16 x)
  k_gemm<<<N_PTS / 128, 256, 0, stream>>>(xb, wbt1, hw);
  k_init_acc<<<N_PTS, 256, 0, stream>>>(hw, invd, acc);
  k_scatter<<<N_PTS, 256, 0, stream>>>(hw, nbr, isd, acc);
  k_finalize_relu<<<N_PTS, 256, 0, stream>>>(acc, b1, hb);
  // layer 2
  k_gemm<<<N_PTS / 128, 256, 0, stream>>>(hb, wbt2, hw);
  k_init_acc<<<N_PTS, 256, 0, stream>>>(hw, invd, acc);
  k_scatter<<<N_PTS, 256, 0, stream>>>(hw, nbr, isd, acc);
  k_finalize_relu<<<N_PTS, 256, 0, stream>>>(acc, b2, hb);
  // layer 3 (no relu, f32 output)
  k_gemm<<<N_PTS / 128, 256, 0, stream>>>(hb, wbt3, hw);
  k_init_acc<<<N_PTS, 256, 0, stream>>>(hw, invd, acc);
  k_scatter<<<N_PTS, 256, 0, stream>>>(hw, nbr, isd, acc);
  k_finalize_out<<<N_PTS, 256, 0, stream>>>(acc, b3, out);
}
